// _UnidirectionalAttention_231928234638
// MI455X (gfx1250) — matrix operand order repaired and checked by reading, not compiled, not run on hardware
//
#include <hip/hip_runtime.h>
#include <hip/hip_bf16.h>

typedef __attribute__((ext_vector_type(16))) _Float16 h16;
typedef __attribute__((ext_vector_type(8)))  float    f8;
typedef __attribute__((ext_vector_type(4)))  unsigned int u4;

// ---------------------------------------------------------------------------
// Fragment helpers (wave32 WMMA layouts per CDNA5 ISA 7.12.2)
//
// A (16x32 f16, row-major source, lda elements):
//   lane l: M = l&15, half = l>>4
//   elems 0..7  -> K = k0 + half*8 + 0..7       (contiguous, 16B)
//   elems 8..15 -> K = k0 + 16 + half*8 + 0..7  (contiguous, 16B)
__device__ __forceinline__ h16 load_frag_a(const _Float16* base, int lda, int row, int k0) {
    int lane = threadIdx.x & 31;
    int m = lane & 15, half = lane >> 4;
    const _Float16* p = base + (row + m) * lda + k0 + half * 8;
    union { u4 u[2]; h16 v; } t;
    t.u[0] = *(const u4*)(p);
    t.u[1] = *(const u4*)(p + 16);
    return t.v;
}

// B (32x16 f16) from an N-major (transposed) source: baseT[n][k], row length ldk.
//   lane l: N = n0 + (l&15); elems 0..7 -> K = k0 + (l>>4)*8 + 0..7, elems 8..15 -> K = k0 + 16 + (l>>4)*8 + 0..7
__device__ __forceinline__ h16 load_frag_b(const _Float16* baseT, int ldk, int n0, int k0) {
    int lane = threadIdx.x & 31;
    int n = lane & 15, half = lane >> 4;
    const _Float16* p = baseT + (n0 + n) * ldk + k0 + half * 8;
    union { u4 u[2]; h16 v; } t;
    t.u[0] = *(const u4*)(p);
    t.u[1] = *(const u4*)(p + 16);
    return t.v;
}

__device__ __forceinline__ f8 wmma32(h16 a, h16 b, f8 c) {
    return __builtin_amdgcn_wmma_f32_16x16x32_f16(false, a, false, b, (short)0, c, false, false);
}

// ---------------------------------------------------------------------------
// Precision staging kernels
__global__ __launch_bounds__(256) void cvt_f16_kernel(const float* src, _Float16* dst, int n) {
    int i = blockIdx.x * blockDim.x + threadIdx.x;
    if (i < n) dst[i] = (_Float16)src[i];
}

// src[rows][cols] (row-major f32) -> dst[cols][rows] (row-major f16)
__global__ __launch_bounds__(256) void cvt_tr_f16_kernel(const float* src, _Float16* dst,
                                                         int rows, int cols) {
    int i = blockIdx.x * blockDim.x + threadIdx.x;
    if (i < rows * cols) {
        int r = i / cols, c = i - r * cols;
        dst[c * rows + r] = (_Float16)src[i];
    }
}

// ---------------------------------------------------------------------------
// QKV projection: C[4096 x 3072] = Xh[4096 x 1024] * WtA^T (+ bias).
// Block: 8 waves, 64 x 256 tile; wave tile 32 x 64 (8 WMMAs / k-step).
// A wave's 64-wide N tile sits entirely inside one Q/K/V section AND one head,
// and its 32-row M tile inside one batch -> routing is wave-uniform (scalar
// branch, no exec-mask divergence).  Q is pre-scaled by 1/sqrt(64); V is
// stored transposed ([b,h,64,tok]) for contiguous PV B-fragments.
__global__ __launch_bounds__(256) void qkv_kernel(const _Float16* __restrict__ Xh,
                                                  const _Float16* __restrict__ WtA,
                                                  const float* __restrict__ bias,
                                                  _Float16* __restrict__ Qb,
                                                  _Float16* __restrict__ Kb,
                                                  _Float16* __restrict__ Vt) {
    int w = threadIdx.x >> 5, lane = threadIdx.x & 31;
    int half = lane >> 4, ln = lane & 15;
    int m0 = blockIdx.y * 64 + (w & 1) * 32;
    int n0 = blockIdx.x * 256 + (w >> 1) * 64;

    f8 acc[2][4] = {};
    for (int k0 = 0; k0 < 1024; k0 += 32) {
        h16 fa0 = load_frag_a(Xh, 1024, m0, k0);
        h16 fa1 = load_frag_a(Xh, 1024, m0 + 16, k0);
        h16 fb[4];
#pragma unroll
        for (int j = 0; j < 4; ++j) fb[j] = load_frag_b(WtA, 1024, n0 + j * 16, k0);
#pragma unroll
        for (int j = 0; j < 4; ++j) {
            acc[0][j] = wmma32(fa0, fb[j], acc[0][j]);
            acc[1][j] = wmma32(fa1, fb[j], acc[1][j]);
        }
    }

    // Wave-uniform routing
    int sec  = n0 >> 10;             // 0=Q, 1=K, 2=V
    int hh   = (n0 & 1023) >> 6;     // head index (uniform: 64-wide tile = 1 head)
    int bb   = m0 >> 11;             // batch (uniform: 64-row tile within a batch)
    int tok0 = m0 & 2047;
    int bh   = bb * 16 + hh;
    float bv[4];
#pragma unroll
    for (int j = 0; j < 4; ++j) bv[j] = bias[n0 + j * 16 + ln];

    if (sec == 0) {
        _Float16* base = Qb + (bh * 2048 + tok0) * 64;
#pragma unroll
        for (int i = 0; i < 2; ++i)
#pragma unroll
            for (int r = 0; r < 8; ++r) {
                int trow = (i * 16 + r + 8 * half) * 64;
#pragma unroll
                for (int j = 0; j < 4; ++j)
                    base[trow + j * 16 + ln] = (_Float16)((acc[i][j][r] + bv[j]) * 0.125f);
            }
    } else if (sec == 1) {
        _Float16* base = Kb + (bh * 2048 + tok0) * 64;
#pragma unroll
        for (int i = 0; i < 2; ++i)
#pragma unroll
            for (int r = 0; r < 8; ++r) {
                int trow = (i * 16 + r + 8 * half) * 64;
#pragma unroll
                for (int j = 0; j < 4; ++j)
                    base[trow + j * 16 + ln] = (_Float16)(acc[i][j][r] + bv[j]);
            }
    } else {
        _Float16* base = Vt + bh * 64 * 2048 + tok0;
#pragma unroll
        for (int j = 0; j < 4; ++j) {
            int drow = (j * 16 + ln) * 2048;
#pragma unroll
            for (int i = 0; i < 2; ++i)
#pragma unroll
                for (int r = 0; r < 8; ++r)
                    base[drow + i * 16 + r + 8 * half] = (_Float16)(acc[i][j][r] + bv[j]);
        }
    }
}

// ---------------------------------------------------------------------------
// Flash attention: one wave per (batch, head, 32-query block).
// K/V fragments are reused across both 16-query tile rows.
__global__ __launch_bounds__(256) void attn_kernel(const _Float16* __restrict__ Qb,
                                                   const _Float16* __restrict__ Kb,
                                                   const _Float16* __restrict__ Vt,
                                                   _Float16* __restrict__ Cb) {
    __shared__ __align__(16) _Float16 lds[8][2][16][32];
    int w = threadIdx.x >> 5, lane = threadIdx.x & 31;
    int half = lane >> 4, ln = lane & 15;
    int gw = blockIdx.x * 8 + w;      // 2048 waves total
    int qb = gw & 63;                 // 64 blocks of 32 queries
    int h  = (gw >> 6) & 15;
    int b  = gw >> 10;
    int q0 = qb * 32;
    int bh = b * 16 + h;

    const _Float16* Qh = Qb + bh * 2048 * 64;
    const _Float16* Kh = Kb + bh * 2048 * 64;
    const _Float16* Vh = Vt + bh * 64 * 2048;

    h16 aq[2][2];
#pragma unroll
    for (int t = 0; t < 2; ++t) {
        aq[t][0] = load_frag_a(Qh, 64, q0 + t * 16, 0);
        aq[t][1] = load_frag_a(Qh, 64, q0 + t * 16, 32);
    }

    f8 o[2][4] = {};
    float mrun[2][8], lrun[2][8];
#pragma unroll
    for (int t = 0; t < 2; ++t)
#pragma unroll
        for (int r = 0; r < 8; ++r) { mrun[t][r] = -1e30f; lrun[t][r] = 0.0f; }

    int nkb = (q0 + 31) >> 5;
    for (int kbi = 0; kbi <= nkb; ++kbi) {
        int kc = kbi * 32;
        // Shared B fragments for this 32-key chunk
        h16 bk0a = load_frag_b(Kh, 64, kc, 0);
        h16 bk0b = load_frag_b(Kh, 64, kc, 32);
        h16 bk1a = load_frag_b(Kh, 64, kc + 16, 0);
        h16 bk1b = load_frag_b(Kh, 64, kc + 16, 32);
        h16 bv[4];
#pragma unroll
        for (int d = 0; d < 4; ++d) bv[d] = load_frag_b(Vh, 2048, d * 16, kc);

        int k0col = kc + ln, k1col = kc + 16 + ln;
#pragma unroll
        for (int t = 0; t < 2; ++t) {
            // S = (Q/8) * K^T : 16 queries x 32 keys
            f8 s0 = {}, s1 = {};
            s0 = wmma32(aq[t][0], bk0a, s0);
            s0 = wmma32(aq[t][1], bk0b, s0);
            s1 = wmma32(aq[t][0], bk1a, s1);
            s1 = wmma32(aq[t][1], bk1b, s1);

            float mnew[8], alpha[8];
#pragma unroll
            for (int r = 0; r < 8; ++r) {
                int q = q0 + t * 16 + r + 8 * half;
                float v0 = s0[r]; if (k0col > q) v0 = -10000.0f;
                float v1 = s1[r]; if (k1col > q) v1 = -10000.0f;
                s0[r] = v0; s1[r] = v1;
                float m = fmaxf(v0, v1);
                for (int x = 1; x < 16; x <<= 1) m = fmaxf(m, __shfl_xor(m, x, 32));
                mnew[r]     = fmaxf(mrun[t][r], m);
                alpha[r]    = __expf(mrun[t][r] - mnew[r]);
                mrun[t][r]  = mnew[r];
            }
#pragma unroll
            for (int r = 0; r < 8; ++r) {
                float p0 = __expf(s0[r] - mnew[r]);
                float p1 = __expf(s1[r] - mnew[r]);
                s0[r] = p0; s1[r] = p1;
                float rs = p0 + p1;
                for (int x = 1; x < 16; x <<= 1) rs += __shfl_xor(rs, x, 32);
                lrun[t][r] = alpha[r] * lrun[t][r] + rs;
#pragma unroll
                for (int d = 0; d < 4; ++d) o[t][d][r] *= alpha[r];
            }
            // Re-layout P: C-form (N=lane, M=elem) -> A-fragment form via wave-private
            // LDS. LDS ops from the same wave are in-order; no barrier needed.
#pragma unroll
            for (int r = 0; r < 8; ++r) {
                int ql = r + 8 * half;
                lds[w][t][ql][ln]      = (_Float16)s0[r];
                lds[w][t][ql][16 + ln] = (_Float16)s1[r];
            }
            h16 pA = load_frag_a(&lds[w][t][0][0], 32, 0, 0);
#pragma unroll
            for (int d = 0; d < 4; ++d) o[t][d] = wmma32(pA, bv[d], o[t][d]);
        }
    }
    // Normalize and write combined [token][h*64+d] as f16 for the final GEMM.
#pragma unroll
    for (int t = 0; t < 2; ++t)
#pragma unroll
        for (int r = 0; r < 8; ++r) {
            float inv = 1.0f / lrun[t][r];
            int row = b * 2048 + q0 + t * 16 + r + 8 * half;
            _Float16* dst = Cb + row * 1024 + h * 64 + ln;
#pragma unroll
            for (int d = 0; d < 4; ++d) dst[d * 16] = (_Float16)(o[t][d][r] * inv);
        }
}

// ---------------------------------------------------------------------------
// Output projection: out[4096 x 1024] = Cb * WtO^T + b_out (fp32 result)
__global__ __launch_bounds__(256) void oproj_kernel(const _Float16* __restrict__ Ch,
                                                    const _Float16* __restrict__ WtO,
                                                    const float* __restrict__ bias,
                                                    float* __restrict__ out) {
    int w = threadIdx.x >> 5, lane = threadIdx.x & 31;
    int half = lane >> 4, ln = lane & 15;
    int m0 = blockIdx.y * 64 + (w & 1) * 32;
    int n0 = blockIdx.x * 256 + (w >> 1) * 64;

    f8 acc[2][4] = {};
    for (int k0 = 0; k0 < 1024; k0 += 32) {
        h16 fa0 = load_frag_a(Ch, 1024, m0, k0);
        h16 fa1 = load_frag_a(Ch, 1024, m0 + 16, k0);
        h16 fb[4];
#pragma unroll
        for (int j = 0; j < 4; ++j) fb[j] = load_frag_b(WtO, 1024, n0 + j * 16, k0);
#pragma unroll
        for (int j = 0; j < 4; ++j) {
            acc[0][j] = wmma32(fa0, fb[j], acc[0][j]);
            acc[1][j] = wmma32(fa1, fb[j], acc[1][j]);
        }
    }
    float bv[4];
#pragma unroll
    for (int j = 0; j < 4; ++j) bv[j] = bias[n0 + j * 16 + ln];
#pragma unroll
    for (int i = 0; i < 2; ++i)
#pragma unroll
        for (int r = 0; r < 8; ++r) {
            int m = m0 + i * 16 + r + 8 * half;
#pragma unroll
            for (int j = 0; j < 4; ++j)
                out[m * 1024 + n0 + j * 16 + ln] = acc[i][j][r] + bv[j];
        }
}

// ---------------------------------------------------------------------------
extern "C" void kernel_launch(void* const* d_in, const int* in_sizes, int n_in,
                              void* d_out, int out_size, void* d_ws, size_t ws_size,
                              hipStream_t stream) {
    (void)in_sizes; (void)n_in; (void)out_size; (void)ws_size;
    const float* X  = (const float*)d_in[0];   // [2,2048,1024]
    const float* Wa = (const float*)d_in[1];   // [1024,3072]
    const float* ba = (const float*)d_in[2];   // [3072]
    const float* Wo = (const float*)d_in[3];   // [1024,1024]
    const float* bo = (const float*)d_in[4];   // [1024]
    float* out = (float*)d_out;                // [2,2048,1024]

    _Float16* p = (_Float16*)d_ws;
    _Float16* Xh  = p; p += 4096 * 1024;        // 8 MB
    _Float16* WtA = p; p += 3072 * 1024;        // 6 MB
    _Float16* WtO = p; p += 1024 * 1024;        // 2 MB
    _Float16* Qb  = p; p += 2 * 16 * 2048 * 64; // 8 MB
    _Float16* Kb  = p; p += 2 * 16 * 2048 * 64; // 8 MB
    _Float16* Vt  = p; p += 2 * 16 * 2048 * 64; // 8 MB
    _Float16* Cb  = p; p += 4096 * 1024;        // 8 MB  (total ~48 MB)

    cvt_f16_kernel<<<(4096 * 1024) / 256, 256, 0, stream>>>(X, Xh, 4096 * 1024);
    cvt_tr_f16_kernel<<<(1024 * 3072) / 256, 256, 0, stream>>>(Wa, WtA, 1024, 3072);
    cvt_tr_f16_kernel<<<(1024 * 1024) / 256, 256, 0, stream>>>(Wo, WtO, 1024, 1024);

    qkv_kernel<<<dim3(12, 64), 256, 0, stream>>>(Xh, WtA, ba, Qb, Kb, Vt);
    attn_kernel<<<256, 256, 0, stream>>>(Qb, Kb, Vt, Cb);
    oproj_kernel<<<dim3(4, 64), 256, 0, stream>>>(Cb, WtO, bo, out);
}
